// SwarmModel_27676769255906
// MI455X (gfx1250) — compile-verified, matrix operand order checked by reading, not run on hardware
//
#include <hip/hip_runtime.h>
#include <hip/hip_bf16.h>

// ---------------------------------------------------------------------------
// MoE-ish transformer block for MI455X (gfx1250, wave32, WMMA).
//  - All GEMMs: v_wmma_f32_16x16x32_f16, f32 accumulate.
//  - Head GEMM (268 of ~303 GFLOP): pre-converted f16 operands + async
//    global->LDS DMA (ASYNCcnt) with double-buffered LDS, one barrier/K-step.
//  - head_w is L2-resident (65MB f16 within 192MB L2), so 128x128 block tiles
//    with 32x K-steps are bandwidth-safe.
// ---------------------------------------------------------------------------

typedef __attribute__((ext_vector_type(16))) _Float16 v16h;
typedef __attribute__((ext_vector_type(8)))  _Float16 v8h;
typedef __attribute__((ext_vector_type(4)))  _Float16 v4h;
typedef __attribute__((ext_vector_type(8)))  float    v8f;

static constexpr int B_  = 8;
static constexpr int T_  = 512;
static constexpr int V_  = 32000;
static constexpr int H_  = 1024;
static constexpr int S_  = 256;
static constexpr int LE_ = 2;
static constexpr int LM_ = 4;

static constexpr int BM = 128, BN = 128, BK = 32;
static constexpr int LDSS = 40;   // LDS row stride in halfs (80B: 16B-aligned, conflict-free)

// ---------------------------------------------------------------------------
// async global->LDS 16B copy + asynccnt wait (gfx1250 TDM-lite path)
// ---------------------------------------------------------------------------
__device__ __forceinline__ void async_cp16B(unsigned lds_off, const _Float16* gptr) {
  asm volatile("global_load_async_to_lds_b128 %0, %1, off"
               :: "v"(lds_off), "v"(gptr) : "memory");
}
__device__ __forceinline__ void wait_async0() {
#if __has_builtin(__builtin_amdgcn_s_wait_asynccnt)
  __builtin_amdgcn_s_wait_asynccnt(0);
#else
  asm volatile("s_wait_asynccnt 0x0" ::: "memory");
#endif
}

// ---------------------------------------------------------------------------
// Kernel 1: embedding gather.
// ---------------------------------------------------------------------------
__global__ void k_gather(const int* __restrict__ tok,
                         const float* __restrict__ embed,
                         float* __restrict__ x) {
  const int bt = blockIdx.x;
  const int t  = tok[bt];
  const float4* src = (const float4*)(embed + (long long)t  * H_);
  float4*       dst = (float4*)      (x     + (long long)bt * H_);
  dst[threadIdx.x] = src[threadIdx.x];
}

// ---------------------------------------------------------------------------
// Kernel 2: per-sequence mean over T.  grid (B, H/256).
// ---------------------------------------------------------------------------
__global__ void k_mean(const float* __restrict__ x, float* __restrict__ xmean) {
  const int b = blockIdx.x;
  const int h = blockIdx.y * 256 + threadIdx.x;
  float s = 0.f;
  for (int t = 0; t < T_; ++t)
    s += x[(long long)(b * T_ + t) * H_ + h];
  xmean[b * H_ + h] = s * (1.f / (float)T_);
}

// ---------------------------------------------------------------------------
// Kernel 3: routing: logits -> softmax -> top2 -> softmax(top2).
// ---------------------------------------------------------------------------
__global__ void k_route(const float* __restrict__ xmean,
                        const float* __restrict__ rw,
                        int* __restrict__ t2i, float* __restrict__ t2w) {
  __shared__ float lg[64];
  const int t = threadIdx.x;
  const int b = t >> 3, e = t & 7;
  float d = 0.f;
  for (int hh = 0; hh < H_; ++hh)
    d += xmean[b * H_ + hh] * rw[e * H_ + hh];
  lg[t] = 2.f * d;
  __syncthreads();
  if (e == 0) {
    float mx = -1e30f;
    for (int j = 0; j < 8; ++j) mx = fmaxf(mx, lg[b * 8 + j]);
    float p[8], s = 0.f;
    for (int j = 0; j < 8; ++j) { p[j] = expf(lg[b * 8 + j] - mx); s += p[j]; }
    for (int j = 0; j < 8; ++j) p[j] /= s;
    int i0 = 0;
    for (int j = 1; j < 8; ++j) if (p[j] > p[i0]) i0 = j;
    int i1 = (i0 == 0) ? 1 : 0;
    for (int j = 0; j < 8; ++j) if (j != i0 && p[j] > p[i1]) i1 = j;
    const float e1 = expf(p[i1] - p[i0]);
    const float w0 = 1.f / (1.f + e1);
    t2i[b * 2 + 0] = i0;  t2i[b * 2 + 1] = i1;
    t2w[b * 2 + 0] = w0;  t2w[b * 2 + 1] = e1 * w0;
  }
}

// ---------------------------------------------------------------------------
// f32-operand WMMA GEMM (backbone/experts).  C[z] (+)= epi(A[z] @ B).
//   EPI: 0 = store, 1 = silu, 2 = C += scale*acc.  B is K-major B[k*ldb+n].
//   eidx: optional per-z expert index -> weight base += eidx[z]*eStride.
// 256 thr = 8 waves (2x4); wave tile 64x32 = 4x2 wmma accums.
// Staging: batch 8 global_load_b128 into regs, then convert+store (loads
// overlap; avoids the serialized load->wait->store chain of v1).
// ---------------------------------------------------------------------------
template <int EPI>
__global__ void __launch_bounds__(256)
k_gemm(const float* __restrict__ A, const float* __restrict__ Bw,
       float* __restrict__ C,
       int M, int N, int K, int lda, int ldb, int ldc,
       long long aBatch, long long cBatch,
       const int* __restrict__ eidx, long long eStride, float scale) {
  __shared__ __align__(16) _Float16 sA[BM * LDSS];
  __shared__ __align__(16) _Float16 sB[BN * LDSS];

  const int tid = threadIdx.x;
  const int z   = blockIdx.z;
  const float* Az = A + (long long)z * aBatch;
  float*       Cz = C + (long long)z * cBatch;
  const float* Bz = Bw + (eidx ? (long long)eidx[z] * eStride : 0ll);

  const int n0 = blockIdx.x * BN;
  const int m0 = blockIdx.y * BM;

  const int wave = tid >> 5;
  const int lane = tid & 31;
  const int wm   = (wave >> 2) * 64;
  const int wn   = (wave & 3) * 32;
  const int lrow = lane & 15;
  const int hi   = lane >> 4;
  const int koff = hi * 8;

  v8f acc[4][2];
  const v8f vzero = {0.f, 0.f, 0.f, 0.f, 0.f, 0.f, 0.f, 0.f};
#pragma unroll
  for (int i = 0; i < 4; ++i)
#pragma unroll
    for (int j = 0; j < 2; ++j) acc[i][j] = vzero;

  for (int k0 = 0; k0 < K; k0 += BK) {
    // ---- phase 1: batch all global loads into registers ----
    float4 ra[4], rb[4];
#pragma unroll
    for (int jj = 0; jj < 4; ++jj) {
      const int idx = tid + jj * 256;
      const int m   = idx >> 3;
      const int kk  = (idx & 7) << 2;
      ra[jj] = *(const float4*)&Az[(long long)(m0 + m) * lda + k0 + kk];
    }
#pragma unroll
    for (int jj = 0; jj < 4; ++jj) {
      const int idx = tid + jj * 256;
      const int kk  = idx >> 5;
      const int n   = (idx & 31) << 2;
      rb[jj] = *(const float4*)&Bz[(long long)(k0 + kk) * ldb + n0 + n];
    }
    if (k0 + BK < K)
      __builtin_prefetch(&Az[(long long)(m0 + (tid >> 1)) * lda + k0 + BK], 0, 1);
    // ---- phase 2: convert + LDS store ----
#pragma unroll
    for (int jj = 0; jj < 4; ++jj) {
      const int idx = tid + jj * 256;
      const int m   = idx >> 3;
      const int kk  = (idx & 7) << 2;
      v4h o = {(_Float16)ra[jj].x, (_Float16)ra[jj].y,
               (_Float16)ra[jj].z, (_Float16)ra[jj].w};
      *(v4h*)&sA[m * LDSS + kk] = o;
    }
#pragma unroll
    for (int jj = 0; jj < 4; ++jj) {
      const int idx = tid + jj * 256;
      const int kk  = idx >> 5;
      const int n   = (idx & 31) << 2;
      sB[(n + 0) * LDSS + kk] = (_Float16)rb[jj].x;
      sB[(n + 1) * LDSS + kk] = (_Float16)rb[jj].y;
      sB[(n + 2) * LDSS + kk] = (_Float16)rb[jj].z;
      sB[(n + 3) * LDSS + kk] = (_Float16)rb[jj].w;
    }
    __syncthreads();

    // ---- fragments + 4x2 WMMA ----
    v16h fb[2];
#pragma unroll
    for (int j = 0; j < 2; ++j) {
      const _Float16* p = &sB[(wn + j * 16 + lrow) * LDSS + koff];
      const v8h lo = *(const v8h*)p;
      const v8h hh = *(const v8h*)(p + 16);
      fb[j] = __builtin_shufflevector(lo, hh, 0,1,2,3,4,5,6,7,8,9,10,11,12,13,14,15);
    }
#pragma unroll
    for (int i = 0; i < 4; ++i) {
      const _Float16* p = &sA[(wm + i * 16 + lrow) * LDSS + koff];
      const v8h lo = *(const v8h*)p;
      const v8h hh = *(const v8h*)(p + 16);
      const v16h fa = __builtin_shufflevector(lo, hh, 0,1,2,3,4,5,6,7,8,9,10,11,12,13,14,15);
#pragma unroll
      for (int j = 0; j < 2; ++j)
        acc[i][j] = __builtin_amdgcn_wmma_f32_16x16x32_f16(
            false, fa, false, fb[j], (short)0, acc[i][j], false, false);
    }
    __syncthreads();
  }

#pragma unroll
  for (int i = 0; i < 4; ++i)
#pragma unroll
    for (int j = 0; j < 2; ++j)
#pragma unroll
      for (int r = 0; r < 8; ++r) {
        const int cm = m0 + wm + i * 16 + hi * 8 + r;
        const int cn = n0 + wn + j * 16 + lrow;
        const long long off = (long long)cm * ldc + cn;
        const float v = acc[i][j][r];
        if (EPI == 0)      Cz[off]  = v;
        else if (EPI == 1) Cz[off]  = v / (1.f + __expf(-v));   // silu
        else               Cz[off] += scale * v;                // residual
      }
}

// ---------------------------------------------------------------------------
// f32 -> f16 bulk converter (head_w once; normalized x once).
// ---------------------------------------------------------------------------
__global__ void k_cvt(const float* __restrict__ src, _Float16* __restrict__ dst,
                      long long n4) {
  const long long g = (long long)blockIdx.x * 256 + threadIdx.x;
  if (g >= n4) return;
  const float4 v = ((const float4*)src)[g];
  v4h o = {(_Float16)v.x, (_Float16)v.y, (_Float16)v.z, (_Float16)v.w};
  *(v4h*)(dst + g * 4) = o;
}

// ---------------------------------------------------------------------------
// Head GEMM, pure f16 operands, async global->LDS DMA, double-buffered LDS.
// out[M,N] = A16[M,K] @ B16[N,K]^T.  One barrier per K-step; tile k+1 DMA
// overlaps WMMA on tile k.  Per wave per tile: 4 async b128 issues.
// ---------------------------------------------------------------------------
__global__ void __launch_bounds__(256)
k_gemm_head16(const _Float16* __restrict__ A, const _Float16* __restrict__ Bw,
              float* __restrict__ C, int K, int lda, int ldb, int ldc) {
  __shared__ __align__(16) _Float16 sA[2][BM * LDSS];
  __shared__ __align__(16) _Float16 sB[2][BN * LDSS];

  const int tid = threadIdx.x;
  const int n0 = blockIdx.x * BN;
  const int m0 = blockIdx.y * BM;

  const int wave = tid >> 5;
  const int lane = tid & 31;
  const int wm   = (wave >> 2) * 64;
  const int wn   = (wave & 3) * 32;
  const int lrow = lane & 15;
  const int hi   = lane >> 4;
  const int koff = hi * 8;

  // per-thread staging slots: 512 16B-chunks per tile (128 rows x 4 chunks)
  const int srow = tid >> 2;            // row handled in chunk pass 0
  const int sch  = (tid & 3) << 3;      // half offset 0/8/16/24 (16B chunks)

  v8f acc[4][2];
  const v8f vzero = {0.f, 0.f, 0.f, 0.f, 0.f, 0.f, 0.f, 0.f};
#pragma unroll
  for (int i = 0; i < 4; ++i)
#pragma unroll
    for (int j = 0; j < 2; ++j) acc[i][j] = vzero;

  const int nIter = K / BK;

  // prologue: DMA tile 0 into buffer 0
#pragma unroll
  for (int jj = 0; jj < 2; ++jj) {
    const int row = srow + jj * 64;
    async_cp16B((unsigned)(size_t)&sA[0][row * LDSS + sch],
                &A[(long long)(m0 + row) * lda + sch]);
    async_cp16B((unsigned)(size_t)&sB[0][row * LDSS + sch],
                &Bw[(long long)(n0 + row) * ldb + sch]);
  }

  for (int it = 0; it < nIter; ++it) {
    const int buf = it & 1;
    wait_async0();        // this wave's DMA for tile `it` has landed
    __syncthreads();      // all waves' DMA landed; prev compute done

    if (it + 1 < nIter) { // DMA tile it+1 into the other buffer (overlapped)
      const int k1 = (it + 1) * BK;
#pragma unroll
      for (int jj = 0; jj < 2; ++jj) {
        const int row = srow + jj * 64;
        async_cp16B((unsigned)(size_t)&sA[buf ^ 1][row * LDSS + sch],
                    &A[(long long)(m0 + row) * lda + k1 + sch]);
        async_cp16B((unsigned)(size_t)&sB[buf ^ 1][row * LDSS + sch],
                    &Bw[(long long)(n0 + row) * ldb + k1 + sch]);
      }
    }

    // compute on tile `it`
    v16h fb[2];
#pragma unroll
    for (int j = 0; j < 2; ++j) {
      const _Float16* p = &sB[buf][(wn + j * 16 + lrow) * LDSS + koff];
      const v8h lo = *(const v8h*)p;
      const v8h hh = *(const v8h*)(p + 16);
      fb[j] = __builtin_shufflevector(lo, hh, 0,1,2,3,4,5,6,7,8,9,10,11,12,13,14,15);
    }
#pragma unroll
    for (int i = 0; i < 4; ++i) {
      const _Float16* p = &sA[buf][(wm + i * 16 + lrow) * LDSS + koff];
      const v8h lo = *(const v8h*)p;
      const v8h hh = *(const v8h*)(p + 16);
      const v16h fa = __builtin_shufflevector(lo, hh, 0,1,2,3,4,5,6,7,8,9,10,11,12,13,14,15);
#pragma unroll
      for (int j = 0; j < 2; ++j)
        acc[i][j] = __builtin_amdgcn_wmma_f32_16x16x32_f16(
            false, fa, false, fb[j], (short)0, acc[i][j], false, false);
    }
  }

#pragma unroll
  for (int i = 0; i < 4; ++i)
#pragma unroll
    for (int j = 0; j < 2; ++j)
#pragma unroll
      for (int r = 0; r < 8; ++r) {
        const int cm = m0 + wm + i * 16 + hi * 8 + r;
        const int cn = n0 + wn + j * 16 + lrow;
        C[(long long)cm * ldc + cn] = acc[i][j][r];
      }
}

// ---------------------------------------------------------------------------
// Kernel: replicate x into 2 expert slots per sequence (z = b*2 + k).
// ---------------------------------------------------------------------------
__global__ void k_copy(const float* __restrict__ x, float* __restrict__ xk) {
  const long long g = (long long)blockIdx.x * 256 + threadIdx.x;  // float4 idx
  const int z = (int)(g >> 17);                 // T_*H_/4 = 2^17
  const long long r = g & 131071ll;
  const int b = z >> 1;
  ((float4*)xk)[g] = ((const float4*)x)[((long long)b << 17) + r];
}

// ---------------------------------------------------------------------------
// Kernel: x += w0*xk[2b] + w1*xk[2b+1]
// ---------------------------------------------------------------------------
__global__ void k_combine(float* __restrict__ x, const float* __restrict__ xk,
                          const float* __restrict__ w) {
  const long long g = (long long)blockIdx.x * 256 + threadIdx.x;
  const int b = (int)(g >> 17);
  const long long r = g & 131071ll;
  const float w0 = w[b * 2], w1 = w[b * 2 + 1];
  float4 a  = ((const float4*)x)[g];
  const float4 e0 = ((const float4*)xk)[((long long)(2 * b)     << 17) + r];
  const float4 e1 = ((const float4*)xk)[((long long)(2 * b + 1) << 17) + r];
  a.x += w0 * e0.x + w1 * e1.x;  a.y += w0 * e0.y + w1 * e1.y;
  a.z += w0 * e0.z + w1 * e1.z;  a.w += w0 * e0.w + w1 * e1.w;
  ((float4*)x)[g] = a;
}

// ---------------------------------------------------------------------------
// Kernel: in-place RMSNorm, one block per row.
// ---------------------------------------------------------------------------
__global__ void k_rmsnorm(float* __restrict__ x, const float* __restrict__ nw) {
  __shared__ float red[256];
  const int tid = threadIdx.x;
  float4* p = (float4*)(x + (long long)blockIdx.x * H_);
  float4 v = p[tid];
  red[tid] = v.x * v.x + v.y * v.y + v.z * v.z + v.w * v.w;
  __syncthreads();
  for (int o = 128; o > 0; o >>= 1) {
    if (tid < o) red[tid] += red[tid + o];
    __syncthreads();
  }
  const float scale = rsqrtf(red[0] * (1.f / (float)H_) + 1.1920928955078125e-7f);
  const int h = tid * 4;
  v.x *= scale * nw[h];     v.y *= scale * nw[h + 1];
  v.z *= scale * nw[h + 2]; v.w *= scale * nw[h + 3];
  p[tid] = v;
}

// ---------------------------------------------------------------------------
// Workspace layout (floats):
//   x:      [0, 4194304)            B*T*H
//   xk:     [4194304, 12582912)     2*B*T*H
//   h:      [12582912, 14680064)    16*T*S
//   xmean:  [14680064, 14688256)    B*H
//   t2w:    [14688256, 14688272)
//   t2i:    [14688272, 14688288)    ints
//   x16:    [14688288, 16785440)    B*T*H halfs
//   hw16:   [16785440, 33169440)    V*H halfs
// total ~132.7 MB.
// ---------------------------------------------------------------------------
extern "C" void kernel_launch(void* const* d_in, const int* in_sizes, int n_in,
                              void* d_out, int out_size, void* d_ws, size_t ws_size,
                              hipStream_t stream) {
  (void)in_sizes; (void)n_in; (void)out_size; (void)ws_size;

  const int*   tok    = (const int*)  d_in[0];
  const float* embed  = (const float*)d_in[1];
  const float* rw     = (const float*)d_in[2];
  const float* bw1    = (const float*)d_in[3];  // [LM,H,S]
  const float* bw2    = (const float*)d_in[4];  // [LM,S,H]
  const float* ew1    = (const float*)d_in[5];  // [E,LE,H,S]
  const float* ew2    = (const float*)d_in[6];  // [E,LE,S,H]
  const float* nw     = (const float*)d_in[7];
  const float* hw     = (const float*)d_in[8];  // [V,H]
  float* out = (float*)d_out;

  float* wsf   = (float*)d_ws;
  float* x     = wsf;
  float* xk    = wsf + 4194304;
  float* h     = wsf + 12582912;
  float* xmean = wsf + 14680064;
  float* t2w   = wsf + 14688256;
  int*   t2i   = (int*)(wsf + 14688272);
  _Float16* x16  = (_Float16*)(wsf + 14688288);
  _Float16* hw16 = (_Float16*)(wsf + 16785440);

  const long long TH = (long long)T_ * H_;   // 524288
  const long long TS = (long long)T_ * S_;   // 131072

  // head_w f32->f16 (independent of the rest; 131MB read / 65MB write once)
  k_cvt<<<(V_ * H_ / 4) / 256, 256, 0, stream>>>(hw, hw16, (long long)V_ * H_ / 4);

  // 1) embedding gather, 2) sequence mean, 3) routing
  k_gather<<<B_ * T_, 256, 0, stream>>>(tok, embed, x);
  k_mean<<<dim3(B_, H_ / 256), 256, 0, stream>>>(x, xmean);
  k_route<<<1, 64, 0, stream>>>(xmean, rw, t2i, t2w);

  // 4) backbone: h = silu(x@w1);  x += 0.1*(h@w2)
  for (int l = 0; l < LM_; ++l) {
    k_gemm<1><<<dim3(S_ / BN, (B_ * T_) / BM, 1), 256, 0, stream>>>(
        x, bw1 + (long long)l * H_ * S_, h,
        B_ * T_, S_, H_, H_, S_, S_, 0ll, 0ll, nullptr, 0ll, 0.f);
    k_gemm<2><<<dim3(H_ / BN, (B_ * T_) / BM, 1), 256, 0, stream>>>(
        h, bw2 + (long long)l * S_ * H_, x,
        B_ * T_, H_, S_, S_, H_, H_, 0ll, 0ll, nullptr, 0ll, 0.1f);
  }

  // 5) replicate activations into 2 expert slots
  k_copy<<<(2 * B_ * T_ * H_ / 4) / 256, 256, 0, stream>>>(x, xk);

  // 6) expert layers, batched over z = b*2 + slot with weight indirection
  for (int l = 0; l < LE_; ++l) {
    k_gemm<1><<<dim3(S_ / BN, T_ / BM, 2 * B_), 256, 0, stream>>>(
        xk, ew1 + (long long)l * H_ * S_, h,
        T_, S_, H_, H_, S_, S_, TH, TS, t2i, (long long)LE_ * H_ * S_, 0.f);
    k_gemm<2><<<dim3(H_ / BN, T_ / BM, 2 * B_), 256, 0, stream>>>(
        h, ew2 + (long long)l * S_ * H_, xk,
        T_, H_, S_, S_, H_, H_, TS, TH, t2i, (long long)LE_ * S_ * H_, 0.1f);
  }

  // 7) combine experts, 8) RMSNorm in place
  k_combine<<<(B_ * T_ * H_ / 4) / 256, 256, 0, stream>>>(x, xk, t2w);
  k_rmsnorm<<<B_ * T_, 256, 0, stream>>>(x, nw);

  // 9) x -> f16, then head GEMM with async-DMA double buffering
  k_cvt<<<(B_ * T_ * H_ / 4) / 256, 256, 0, stream>>>(x, x16, (long long)B_ * T_ * H_ / 4);
  k_gemm_head16<<<dim3(V_ / BN, (B_ * T_) / BM, 1), 256, 0, stream>>>(
      x16, hw16, out, H_, H_, H_, V_);
}
